// PlainGCN_14353780703616
// MI455X (gfx1250) — matrix operand order repaired and checked by reading, not compiled, not run on hardware
//
#include <hip/hip_runtime.h>

#define N_PTS      16384
#define KNN_K      16
#define CH         64
#define BATCH_PTS  4096

typedef __attribute__((ext_vector_type(16))) _Float16 v16h;
typedef __attribute__((ext_vector_type(8)))  float    v8f;

// ---------------------------------------------------------------------------
// Weight prep: W is (128 x 64) row-major.  EdgeConv factorization:
//   e@W = x_i @ (Wtop - Wbot) + x_j @ Wbot
// Store U = Wtop - Wbot and V = Wbot as f16, TRANSPOSED to (n, k) layout so
// WMMA B-fragments load as contiguous 16B chunks per lane.
// ---------------------------------------------------------------------------
__global__ void prep_weights(const float* __restrict__ W,
                             _Float16* __restrict__ Ut,   // [n*64 + k]
                             _Float16* __restrict__ Vt)   // [n*64 + k]
{
    int t = blockIdx.x * blockDim.x + threadIdx.x;
    if (t >= CH * CH) return;
    int k = t >> 6;
    int n = t & 63;
    float top = W[k * CH + n];
    float bot = W[(k + CH) * CH + n];
    Ut[n * CH + k] = (_Float16)(top - bot);
    Vt[n * CH + k] = (_Float16)bot;
}

// f32 -> f16 row-major convert (for WMMA A-matrix source)
__global__ void f32_to_f16(const float* __restrict__ in, _Float16* __restrict__ out, int n)
{
    int t = blockIdx.x * blockDim.x + threadIdx.x;
    if (t < n) out[t] = (_Float16)in[t];
}

// ---------------------------------------------------------------------------
// KNN: one wave32 per query point.  Candidates of the query's batch (4096
// points, 48KB of xyz) are staged in LDS.  Exact integer distances; key packs
// (dist<<12)|local_idx into u32 (dist <= 3*431^2 < 2^20).  Each lane keeps a
// sorted top-16 in registers over a lane-strided scan, then 16 rounds of
// wave-min merge via shuffles.
// ---------------------------------------------------------------------------
__global__ void __launch_bounds__(256) knn_kernel(const int* __restrict__ vc,
                                                  int* __restrict__ knn_idx)
{
    __shared__ int spos[BATCH_PTS * 3];   // 48 KB (of 320 KB WGP LDS)

    const int lane = threadIdx.x & 31;
    const int wave = threadIdx.x >> 5;
    const int q    = blockIdx.x * 8 + wave;   // 8 waves/block, all same batch
    const int base = (q >> 12) << 12;         // batch start (4096-aligned)

    const int4* vc4 = (const int4*)vc;        // (batch, x, y, z)
    for (int p = threadIdx.x; p < BATCH_PTS; p += 256) {
        int4 v = vc4[base + p];
        spos[3 * p + 0] = v.y;
        spos[3 * p + 1] = v.z;
        spos[3 * p + 2] = v.w;
    }
    __syncthreads();

    int4 qv = vc4[q];
    const int qx = qv.y, qy = qv.z, qz = qv.w;
    const int qloc = q - base;

    unsigned arr[KNN_K];
    #pragma unroll
    for (int i = 0; i < KNN_K; ++i) arr[i] = 0xFFFFFFFFu;

    for (int j = lane; j < BATCH_PTS; j += 32) {
        int dx = spos[3 * j + 0] - qx;
        int dy = spos[3 * j + 1] - qy;
        int dz = spos[3 * j + 2] - qz;
        unsigned dist = (unsigned)(dx * dx + dy * dy + dz * dz);
        unsigned key  = (dist << 12) | (unsigned)j;
        if (j != qloc && key < arr[KNN_K - 1]) {
            arr[KNN_K - 1] = key;
            #pragma unroll
            for (int t = KNN_K - 1; t > 0; --t) {
                if (arr[t] < arr[t - 1]) {
                    unsigned tmp = arr[t - 1]; arr[t - 1] = arr[t]; arr[t] = tmp;
                }
            }
        }
    }

    // Merge the 32 sorted per-lane lists: 16x (wave-min, winner pops head).
    #pragma unroll 1
    for (int t = 0; t < KNN_K; ++t) {
        unsigned m = arr[0];
        #pragma unroll
        for (int s = 16; s >= 1; s >>= 1) {
            unsigned o = (unsigned)__shfl_xor((int)m, s, 32);
            m = (o < m) ? o : m;
        }
        if (lane == 0) knn_idx[q * KNN_K + t] = base + (int)(m & 0xFFFu);
        if (arr[0] == m) {            // keys are unique -> exactly one winner
            #pragma unroll
            for (int i = 0; i < KNN_K - 1; ++i) arr[i] = arr[i + 1];
            arr[KNN_K - 1] = 0xFFFFFFFFu;
        }
    }
}

// ---------------------------------------------------------------------------
// GEMM: Y(M x 64) = f16(X) @ Wt^T (+ bias), Wt stored (n,k).  One wave per
// 16-row strip; 4 N-tiles x 2 K-blocks of v_wmma_f32_16x16x32_f16.
// A layout (ISA 16-bit A 16x32): lanes 0-15 -> M=lane, K halves {kb..kb+7,
// kb+16..kb+23}; lanes 16-31 -> +8.  B layout: same K order as A,
// N = lane&15.
// ---------------------------------------------------------------------------
__global__ void __launch_bounds__(256) gemm_x64(const _Float16* __restrict__ Xh,
                                                const _Float16* __restrict__ Wt,
                                                const float* __restrict__ bias, // may be null
                                                float* __restrict__ Y)
{
    const int lane = threadIdx.x & 31;
    const int wave = threadIdx.x >> 5;
    const int m0   = (blockIdx.x * 8 + wave) * 16;
    const int lh   = lane >> 4;
    const int l15  = lane & 15;
    const int row  = m0 + l15;

    v8f acc[4];
    const v8f vzero = {0.f, 0.f, 0.f, 0.f, 0.f, 0.f, 0.f, 0.f};
    #pragma unroll
    for (int nb = 0; nb < 4; ++nb) acc[nb] = vzero;

    #pragma unroll
    for (int kb = 0; kb < CH; kb += 32) {
        // --- A fragment ---
        const int koff = kb + lh * 8;
        const _Float16* ap = &Xh[row * CH + koff];
        v16h a;
        #pragma unroll
        for (int j = 0; j < 8; ++j) a[j] = ap[j];            // K = koff..koff+7
        #pragma unroll
        for (int j = 0; j < 8; ++j) a[8 + j] = ap[16 + j];   // K = koff+16..koff+23

        #pragma unroll
        for (int nb = 0; nb < 4; ++nb) {
            // --- B fragment ---
            const int col = nb * 16 + l15;
            const _Float16* bp = &Wt[col * CH + kb + lh * 8];
            v16h b;
            #pragma unroll
            for (int j = 0; j < 8; ++j) b[j] = bp[j];        // K = kb+8h..+7
            #pragma unroll
            for (int j = 0; j < 8; ++j) b[8 + j] = bp[16 + j]; // K = kb+16+8h..+7
            acc[nb] = __builtin_amdgcn_wmma_f32_16x16x32_f16(
                false, a, false, b, (short)0, acc[nb], false, false);
        }
    }

    #pragma unroll
    for (int nb = 0; nb < 4; ++nb) {
        const int col = nb * 16 + l15;
        const float bv = bias ? bias[col] : 0.f;
        #pragma unroll
        for (int r = 0; r < 8; ++r) {
            const int orow = m0 + r + lh * 8;   // D: VGPR r -> M=r / M=r+8
            Y[orow * CH + col] = acc[nb][r] + bv;
        }
    }
}

// ---------------------------------------------------------------------------
// Edge reduce: out[n,c] = max_k relu(P[n,c] + Q[idx[n,k],c]) (+ residual).
// Q rows are L2-resident (4 MB per layer vs 192 MB L2).
// ---------------------------------------------------------------------------
__global__ void __launch_bounds__(256) edge_max_kernel(const float* __restrict__ P,
                                                       const float* __restrict__ Q,
                                                       const int* __restrict__ knn_idx,
                                                       const float* __restrict__ residual, // may be null
                                                       float* __restrict__ out)
{
    int t = blockIdx.x * blockDim.x + threadIdx.x;   // N*64 threads
    int n = t >> 6;
    int c = t & 63;
    float p = P[t];
    float m = 0.f;                                   // relu outputs are >= 0
    #pragma unroll
    for (int k = 0; k < KNN_K; ++k) {
        int j = knn_idx[n * KNN_K + k];
        float h = p + Q[j * CH + c];
        h = h > 0.f ? h : 0.f;
        m = h > m ? h : m;
    }
    out[t] = residual ? (m + residual[t]) : m;
}

// ---------------------------------------------------------------------------
extern "C" void kernel_launch(void* const* d_in, const int* in_sizes, int n_in,
                              void* d_out, int out_size, void* d_ws, size_t ws_size,
                              hipStream_t stream)
{
    const int*   vc = (const int*)  d_in[0];   // voxel_coords (N,4) int32
    const float* x  = (const float*)d_in[1];   // pillar_features (N,64) f32
    const float* W1 = (const float*)d_in[2];   // (128,64)
    const float* b1 = (const float*)d_in[3];   // (64,)
    const float* W2 = (const float*)d_in[4];   // (128,64)
    const float* b2 = (const float*)d_in[5];   // (64,)
    float* out = (float*)d_out;                // (N,64) f32

    char* ws = (char*)d_ws;
    size_t off = 0;
    int* knn_idx = (int*)(ws + off);        off += (size_t)N_PTS * KNN_K * sizeof(int);
    _Float16* U1 = (_Float16*)(ws + off);   off += (size_t)CH * CH * sizeof(_Float16);
    _Float16* V1 = (_Float16*)(ws + off);   off += (size_t)CH * CH * sizeof(_Float16);
    _Float16* U2 = (_Float16*)(ws + off);   off += (size_t)CH * CH * sizeof(_Float16);
    _Float16* V2 = (_Float16*)(ws + off);   off += (size_t)CH * CH * sizeof(_Float16);
    off = (off + 255) & ~(size_t)255;
    float* P  = (float*)(ws + off);         off += (size_t)N_PTS * CH * sizeof(float);
    float* Q  = (float*)(ws + off);         off += (size_t)N_PTS * CH * sizeof(float);
    float* F  = (float*)(ws + off);         off += (size_t)N_PTS * CH * sizeof(float);
    _Float16* Xh = (_Float16*)(ws + off);   off += (size_t)N_PTS * CH * sizeof(_Float16);

    const int NE = N_PTS * CH;                 // 1,048,576 elements

    // weights
    prep_weights<<<16, 256, 0, stream>>>(W1, U1, V1);
    prep_weights<<<16, 256, 0, stream>>>(W2, U2, V2);

    // knn (one wave per query)
    knn_kernel<<<N_PTS / 8, 256, 0, stream>>>(vc, knn_idx);

    // ---- layer 1 ----
    f32_to_f16<<<NE / 256, 256, 0, stream>>>(x, Xh, NE);
    gemm_x64<<<128, 256, 0, stream>>>(Xh, U1, b1, P);
    gemm_x64<<<128, 256, 0, stream>>>(Xh, V1, nullptr, Q);
    edge_max_kernel<<<NE / 256, 256, 0, stream>>>(P, Q, knn_idx, nullptr, F);

    // ---- layer 2 (+ residual) ----
    f32_to_f16<<<NE / 256, 256, 0, stream>>>(F, Xh, NE);
    gemm_x64<<<128, 256, 0, stream>>>(Xh, U2, b2, P);
    gemm_x64<<<128, 256, 0, stream>>>(Xh, V2, nullptr, Q);
    edge_max_kernel<<<NE / 256, 256, 0, stream>>>(P, Q, knn_idx, x, out);
}
